// RNN_79216376808164
// MI455X (gfx1250) — matrix operand order repaired and checked by reading, compiled, not run on hardware
//
#include <hip/hip_runtime.h>
#include <hip/hip_bf16.h>
#include <math.h>

// ---------------------------------------------------------------------------
// Fused RNN for MI455X (gfx1250, wave32).
//   f   = emb[w]                      (B,T,128)
//   fin = f @ W_f^T + b_in            (B,T,512)
//   s_t = tanh(fin_t + s_{t-1} @ W_s^T)      <- sequential over T=256
//   y   = s @ W_out^T + b_out         (B,T,128)
// Batch=256 tiled 16-wide (WMMA M) over 16 workgroups; state lives in LDS,
// weights pre-packed to bf16 WMMA B-operand layout and streamed from L2.
// ---------------------------------------------------------------------------

typedef __bf16 v16bf __attribute__((ext_vector_type(16)));
typedef float  v8f   __attribute__((ext_vector_type(8)));

#define NEMB    128
#define NSTATE  512
#define NALPHA  128
#define SEQ_T   256
#define BATCH   256
#define BT      16          // batch tile = WMMA M
#define THREADS 256         // 8 waves (wave32)

#define SROW 520            // bf16 elems per state row (512 + 8 pad, 1040 B)
#define FROW 136            // bf16 elems per f row    (128 + 8 pad,  272 B)

union Frag {
    uint4 u[2];             // 2 x b128
    v16bf v;                // 16 bf16 per lane
};

__device__ inline unsigned lds_off(const void* p) {
    // generic -> LDS addrspace(3) -> 32-bit LDS offset (what async-to-LDS wants)
    return (unsigned)(unsigned long long)(__attribute__((address_space(3))) const void*)p;
}

// ---------------------------------------------------------------------------
// Prep: f32 -> bf16 copy
// ---------------------------------------------------------------------------
__global__ void k_cvt_bf16(const float* __restrict__ src, __bf16* __restrict__ dst, int n) {
    int i = blockIdx.x * 256 + threadIdx.x;
    if (i < n) dst[i] = (__bf16)src[i];
}

// ---------------------------------------------------------------------------
// Prep: pack W^T into WMMA bf16 B-operand blocks.
// B[k][n] = src[n*srcStride + colOff + k],  n in [0,N), k in [0,K).
// Block = one 32(K) x 16(N) tile = 1024 B; lane L holds 16 bf16 (32 B):
//   n = ntile*16 + (L&15),  k = kblk*32 + (L>>4)*16 + j   (j = 0..15)
// blk = ntile*(K/32) + kblk; lane data contiguous -> hot loop is 2x b128/lane.
// ---------------------------------------------------------------------------
__global__ void k_pack_b(const float* __restrict__ src, int srcStride, int colOff,
                         int K, __bf16* __restrict__ dst, int total) {
    int i = blockIdx.x * 256 + threadIdx.x;
    if (i >= total) return;
    int blk  = i >> 9;          // 512 elems / block
    int rem  = i & 511;
    int lane = rem >> 4;
    int j    = rem & 15;
    int kblocks = K >> 5;
    int ntile = blk / kblocks;
    int kblk  = blk - ntile * kblocks;
    int n = ntile * 16 + (lane & 15);
    int k = kblk * 32 + ((lane >> 4) << 4) + j;
    dst[i] = (__bf16)src[n * srcStride + colOff + k];
}

// ---------------------------------------------------------------------------
// Fused scan kernel: 16 blocks x 256 threads.
// ---------------------------------------------------------------------------
__global__ __launch_bounds__(THREADS, 1)
void rnn_fused(const int*   __restrict__ w,
               const float* __restrict__ b_in,
               const float* __restrict__ b_out,
               const __bf16* __restrict__ embb,
               const __bf16* __restrict__ wfp,   // packed W_f^T   (K=128,N=512)
               const __bf16* __restrict__ wsp,   // packed W_s^T   (K=512,N=512)
               const __bf16* __restrict__ wop,   // packed W_out^T (K=512,N=128)
               float* __restrict__ out)
{
    __shared__ __bf16 sh_state[2][BT * SROW];   // double-buffered state tile
    __shared__ __bf16 sh_f[2][BT * FROW];       // double-buffered embedded tokens

    const int tid   = threadIdx.x;
    const int lane  = tid & 31;
    const int wv    = tid >> 5;        // wave id 0..7
    const int btile = blockIdx.x;
    const int hi    = lane >> 4;       // half-wave select for A/C layouts
    const int lnm   = lane & 15;

    // state0 = 0
    for (int i = tid; i < BT * SROW; i += THREADS) sh_state[0][i] = (__bf16)0.0f;

    // Async stage of f(t): thread -> row m = tid>>4, 8 elems at e0=(tid&15)*8.
    const int frow = tid >> 4;
    const int fe0  = (tid & 15) * 8;
    auto stage_f = [&](int t, int buf) {
        int token = w[(btile * BT + frow) * SEQ_T + t];
        unsigned long long gaddr =
            (unsigned long long)(const void*)(embb + token * NEMB + fe0);
        unsigned laddr = lds_off(&sh_f[buf][frow * FROW + fe0]);
        asm volatile("global_load_async_to_lds_b128 %0, %1, off"
                     :: "v"(laddr), "v"(gaddr) : "memory");
    };

    // A-operand gather from LDS: 16x32 bf16 tile, per-lane layout (matches packed B):
    //  lanes 0-15 : M=lane,    K = kb*32 + {0..15}
    //  lanes 16-31: M=lane-16, K = kb*32 + {16..31}
    auto load_a = [&](const __bf16* base, int rowBytes, int kb) {
        Frag f;
        const char* p = (const char*)base + lnm * rowBytes + kb * 64 + hi * 32;
        f.u[0] = *(const uint4*)p;
        f.u[1] = *(const uint4*)(p + 16);
        return f;
    };
    // B-operand: packed blocks, 32 contiguous bytes per lane.
    auto load_b = [&](const __bf16* pack, int blk) {
        Frag f;
        const char* p = (const char*)pack + blk * 1024 + lane * 32;
        f.u[0] = *(const uint4*)p;
        f.u[1] = *(const uint4*)(p + 16);
        return f;
    };

    // Prologue: stage f(0) into buffer 0.
    stage_f(0, 0);
    asm volatile("s_wait_asynccnt 0x0" ::: "memory");
    __syncthreads();

    int sbuf = 0, fbuf = 0;
    for (int t = 0; t < SEQ_T; ++t) {
        // Prefetch next step's embedded tokens while we compute.
        int tn = (t + 1 < SEQ_T) ? (t + 1) : t;
        stage_f(tn, fbuf ^ 1);

        const __bf16* st_old = sh_state[sbuf];
        __bf16*       st_new = sh_state[sbuf ^ 1];
        const __bf16* fbase  = sh_f[fbuf];

        // --- fin + recurrence: wave owns 4 N-tiles (64 states) ---
        for (int nt = 0; nt < 4; ++nt) {
            const int ntile = wv * 4 + nt;
            float bias = b_in[ntile * 16 + lnm];
            v8f c;
            #pragma unroll
            for (int r = 0; r < 8; ++r) c[r] = bias;

            #pragma unroll
            for (int kb = 0; kb < NEMB / 32; ++kb) {        // fin: K=128
                Frag a = load_a(fbase, FROW * 2, kb);
                Frag b = load_b(wfp, ntile * (NEMB / 32) + kb);
                c = __builtin_amdgcn_wmma_f32_16x16x32_bf16(
                        false, a.v, false, b.v, (short)0, c, false, false);
            }
            #pragma unroll
            for (int kb = 0; kb < NSTATE / 32; ++kb) {      // recurrence: K=512
                Frag a = load_a(st_old, SROW * 2, kb);
                Frag b = load_b(wsp, ntile * (NSTATE / 32) + kb);
                c = __builtin_amdgcn_wmma_f32_16x16x32_bf16(
                        false, a.v, false, b.v, (short)0, c, false, false);
            }
            // tanh -> new state buffer (C/D layout: VGPR r, M = r + hi*8, N = lane%16)
            #pragma unroll
            for (int r = 0; r < 8; ++r) {
                int m = r + hi * 8;
                st_new[m * SROW + ntile * 16 + lnm] = (__bf16)tanhf(c[r]);
            }
        }
        __syncthreads();   // full new state visible

        // --- output projection: wave wv owns alphabet tile atile = wv ---
        {
            const int atile = wv;
            float bias = b_out[atile * 16 + lnm];
            v8f c;
            #pragma unroll
            for (int r = 0; r < 8; ++r) c[r] = bias;
            #pragma unroll
            for (int kb = 0; kb < NSTATE / 32; ++kb) {
                Frag a = load_a(st_new, SROW * 2, kb);
                Frag b = load_b(wop, atile * (NSTATE / 32) + kb);
                c = __builtin_amdgcn_wmma_f32_16x16x32_bf16(
                        false, a.v, false, b.v, (short)0, c, false, false);
            }
            #pragma unroll
            for (int r = 0; r < 8; ++r) {
                int m = r + hi * 8;
                size_t idx = ((size_t)(btile * BT + m) * SEQ_T + t) * NALPHA
                             + atile * 16 + lnm;
                out[idx] = c[r];
            }
        }

        asm volatile("s_wait_asynccnt 0x0" ::: "memory");  // f(t+1) landed (this wave)
        __syncthreads();                                   // ... and all waves'
        sbuf ^= 1;
        fbuf ^= 1;
    }
}

// ---------------------------------------------------------------------------
extern "C" void kernel_launch(void* const* d_in, const int* in_sizes, int n_in,
                              void* d_out, int out_size, void* d_ws, size_t ws_size,
                              hipStream_t stream) {
    const int*   w     = (const int*)d_in[0];
    const float* emb   = (const float*)d_in[1];
    const float* W_in  = (const float*)d_in[2];   // (512, 640)
    const float* b_in  = (const float*)d_in[3];
    const float* W_out = (const float*)d_in[4];   // (128, 512)
    const float* b_out = (const float*)d_in[5];
    float* out = (float*)d_out;

    char* ws = (char*)d_ws;                       // ~800 KB used
    __bf16* embb = (__bf16*)(ws);                             // 32 KB
    __bf16* wfp  = (__bf16*)(ws + 32768);                     // 128 KB
    __bf16* wsp  = (__bf16*)(ws + 32768 + 131072);            // 512 KB
    __bf16* wop  = (__bf16*)(ws + 32768 + 131072 + 524288);   // 128 KB

    k_cvt_bf16<<<(NALPHA * NEMB + 255) / 256, 256, 0, stream>>>(emb, embb, NALPHA * NEMB);
    // W_f^T: N=512 outputs, K=128, cols [0,128) of W_in (stride 640)
    k_pack_b<<<(NSTATE * NEMB) / 256, 256, 0, stream>>>(
        W_in, NEMB + NSTATE, 0, NEMB, wfp, NSTATE * NEMB);
    // W_s^T: N=512, K=512, cols [128,640)
    k_pack_b<<<(NSTATE * NSTATE) / 256, 256, 0, stream>>>(
        W_in, NEMB + NSTATE, NEMB, NSTATE, wsp, NSTATE * NSTATE);
    // W_out^T: N=128, K=512
    k_pack_b<<<(NALPHA * NSTATE) / 256, 256, 0, stream>>>(
        W_out, NSTATE, 0, NSTATE, wop, NALPHA * NSTATE);

    rnn_fused<<<BATCH / BT, THREADS, 0, stream>>>(
        w, b_in, b_out, embb, wfp, wsp, wop, out);
}
